// VAE_RC_90477781057608
// MI455X (gfx1250) — matrix operand order repaired and checked by reading, not compiled, not run on hardware
//
#include <hip/hip_runtime.h>
#include <hip/hip_bf16.h>

// ---------------------------------------------------------------------------
// Types for CDNA5 WMMA / TDM (gfx1250, wave32)
// ---------------------------------------------------------------------------
typedef __attribute__((ext_vector_type(16))) __bf16        v16bf;
typedef __attribute__((ext_vector_type(8)))  __bf16        v8bf;
typedef __attribute__((ext_vector_type(8)))  float         v8f;
typedef __attribute__((ext_vector_type(4)))  float         v4f;
typedef __attribute__((ext_vector_type(4)))  unsigned int  u32x4;
typedef __attribute__((ext_vector_type(8)))  int           i32x8;
typedef __attribute__((ext_vector_type(4)))  int           i32x4;

#define T_DIM 8192
#define D_DIM 1024
#define H_DIM 512
#define N_DIM 1024
#define C_DIM 10

// Toolchain discriminator for __builtin_amdgcn_tensor_load_to_lds arity:
// the gfx1250 TDM header ships only with the 6-arg (therock) toolchain.
#if __has_include(<hip/amd_detail/amd_gfx1250_TDM.h>)
#define TDM_ARITY6 1
#endif

// ---------------------------------------------------------------------------
// Tensor Data Mover: 2D tile load Global -> LDS (D# per CDNA5 ISA ch.8).
// dim0 = row length (elements), dim1 = #rows, stride0 = row stride (elements),
// 4-byte elements, tile == tensor (full-slice copy), no multicast/pad/iterate.
// ---------------------------------------------------------------------------
#if __has_builtin(__builtin_amdgcn_tensor_load_to_lds)
__device__ __forceinline__ void tdm_load_tile_2d_f32(
    unsigned lds_byte_off, const float* gsrc,
    unsigned dim0, unsigned dim1, unsigned stride0)
{
    unsigned long long ga = (unsigned long long)(const void*)gsrc;
    u32x4 g0;
    g0[0] = 1u;                                       // count=1, user descriptor
    g0[1] = lds_byte_off;                             // lds_addr
    g0[2] = (unsigned)ga;                             // global_addr[31:0]
    g0[3] = (unsigned)((ga >> 32) & 0x01FFFFFFu)      // global_addr[56:32]
          | (2u << 30);                               // type=2 ("image")
    i32x8 g1;
    g1[0] = (int)(2u << 16);                          // wg_mask=0, data_size=4B
    g1[1] = (int)((dim0 & 0xFFFFu) << 16);            // tensor_dim0[15:0]
    g1[2] = (int)(((dim0 >> 16) & 0xFFFFu)            // tensor_dim0[31:16]
          | ((dim1 & 0xFFFFu) << 16));                // tensor_dim1[15:0]
    g1[3] = (int)(((dim1 >> 16) & 0xFFFFu)            // tensor_dim1[31:16]
          | ((dim0 & 0xFFFFu) << 16));                // tile_dim0 = dim0
    g1[4] = (int)(dim1 & 0xFFFFu);                    // tile_dim1 = dim1, tile_dim2=0
    g1[5] = (int)stride0;                             // tensor_dim0_stride[31:0]
    g1[6] = 0;                                        // stride hi / dim1_stride lo
    g1[7] = 0;                                        // dim1_stride hi
    i32x4 z4 = {0, 0, 0, 0};
#if defined(TDM_ARITY6)
    i32x8 z8 = {0, 0, 0, 0, 0, 0, 0, 0};
    __builtin_amdgcn_tensor_load_to_lds(g0, g1, z4, z4, z8, 0);
#else
    __builtin_amdgcn_tensor_load_to_lds(g0, g1, z4, z4, 0);
#endif
}
#endif

// ===========================================================================
// GEMM:  C[M x N] = act( A[M x K] @ W[N x K]^T + bias )
// 256 threads = 8 waves; block tile 128(M) x 64(N); K-stage 64.
// Register->LDS ping-pong pipeline: stage kt+1 global loads overlap the
// 8 v_wmma_f32_16x16x32_bf16 issued from LDS buffer p; ONE barrier per stage.
// Fragment gathers: row/col = lane%16, k = 16*(lane/16) + s for both A and B.
// NGUARD=0 compiles the guard-free path (N multiple of 64).
// ===========================================================================
template<int ACT, int HAS_BIAS, int NGUARD>
__global__ __launch_bounds__(256) void gemm_wmma(
    const float* __restrict__ A, const float* __restrict__ W,
    const float* __restrict__ bias, float* __restrict__ C,
    int M, int N, int K)
{
    __shared__ __align__(16) __bf16 As[2][128][72];   // 64 + 8 pad (144B rows)
    __shared__ __align__(16) __bf16 Bs[2][64][72];

    const int tid  = threadIdx.x;
    const int lane = tid & 31;
    const int wave = tid >> 5;
    const int wm   = (wave & 3) << 5;      // wave M offset in block tile
    const int wn   = (wave >> 2) << 5;     // wave N offset in block tile
    const int blockM = blockIdx.x * 128;
    const int blockN = blockIdx.y * 64;

    const int  mlo = lane & 15;
    const bool hi  = (lane >= 16);

    v8f acc[2][2] = {};

    // global->register staging maps (K-stage = 64)
    const int arow = tid >> 1;             // 0..127
    const int acol = (tid & 1) << 5;       // 0 or 32
    const int brow = tid >> 2;             // 0..63
    const int bcol = (tid & 3) << 4;       // 0,16,32,48
    const int wrowN = blockN + brow;
    const bool bvalid = (!NGUARD) || (wrowN < N);

    v4f areg[8], breg[4];

    auto load_tile = [&](int k0) {
        const float* ap = A + (size_t)(blockM + arow) * K + (k0 + acol);
        #pragma unroll
        for (int i = 0; i < 8; ++i)
            areg[i] = *reinterpret_cast<const v4f*>(ap + i * 4);
        if (bvalid) {
            const float* bp = W + (size_t)wrowN * K + (k0 + bcol);
            #pragma unroll
            for (int i = 0; i < 4; ++i)
                breg[i] = *reinterpret_cast<const v4f*>(bp + i * 4);
        } else {
            #pragma unroll
            for (int i = 0; i < 4; ++i) breg[i] = v4f{0.f, 0.f, 0.f, 0.f};
        }
    };
    auto store_tile = [&](int p) {
        #pragma unroll
        for (int i = 0; i < 8; ++i) {
            As[p][arow][acol + i * 4 + 0] = (__bf16)areg[i].x;
            As[p][arow][acol + i * 4 + 1] = (__bf16)areg[i].y;
            As[p][arow][acol + i * 4 + 2] = (__bf16)areg[i].z;
            As[p][arow][acol + i * 4 + 3] = (__bf16)areg[i].w;
        }
        #pragma unroll
        for (int i = 0; i < 4; ++i) {
            Bs[p][brow][bcol + i * 4 + 0] = (__bf16)breg[i].x;
            Bs[p][brow][bcol + i * 4 + 1] = (__bf16)breg[i].y;
            Bs[p][brow][bcol + i * 4 + 2] = (__bf16)breg[i].z;
            Bs[p][brow][bcol + i * 4 + 3] = (__bf16)breg[i].w;
        }
    };

    const int nk = K >> 6;                 // K is a multiple of 64 here
    load_tile(0);
    store_tile(0);
    __syncthreads();

    int p = 0;
    for (int kt = 0; kt < nk; ++kt) {
        const bool more = (kt + 1 < nk);
        if (more) {
            load_tile((kt + 1) << 6);      // overlap with WMMA below
            // L2 prefetch two stages ahead (global_prefetch_b8)
            if (kt + 2 < nk)
                __builtin_prefetch(
                    A + (size_t)(blockM + arow) * K + ((kt + 2) << 6) + acol, 0, 1);
        }
        // ---- 8 WMMAs out of LDS buffer p ----
        #pragma unroll
        for (int kk = 0; kk < 64; kk += 32) {
            union { v16bf v; v8bf h[2]; } af[2], bf[2];
            #pragma unroll
            for (int i = 0; i < 2; ++i) {
                const int m = wm + i * 16 + mlo;
                af[i].h[0] = *reinterpret_cast<const v8bf*>(&As[p][m][kk + (hi ? 16 : 0)]);
                af[i].h[1] = *reinterpret_cast<const v8bf*>(&As[p][m][kk + (hi ? 24 : 8)]);
            }
            #pragma unroll
            for (int j = 0; j < 2; ++j) {
                const int n  = wn + j * 16 + mlo;
                const int kb = kk + (hi ? 16 : 0);
                bf[j].h[0] = *reinterpret_cast<const v8bf*>(&Bs[p][n][kb]);
                bf[j].h[1] = *reinterpret_cast<const v8bf*>(&Bs[p][n][kb + 8]);
            }
            #pragma unroll
            for (int i = 0; i < 2; ++i)
                #pragma unroll
                for (int j = 0; j < 2; ++j)
                    acc[i][j] = __builtin_amdgcn_wmma_f32_16x16x32_bf16(
                        false, af[i].v, false, bf[j].v,
                        (short)0, acc[i][j], false, false);
        }
        if (more) store_tile(p ^ 1);       // fill the other buffer
        __syncthreads();
        p ^= 1;
    }

    // ---- epilogue: bias + activation + store (C/D layout per ISA 7.12.2) ----
    #pragma unroll
    for (int j = 0; j < 2; ++j) {
        const int col = blockN + wn + j * 16 + mlo;
        const bool cok = (!NGUARD) || (col < N);
        float bv = 0.0f;
        if (HAS_BIAS && cok) bv = bias[col];
        #pragma unroll
        for (int i = 0; i < 2; ++i) {
            #pragma unroll
            for (int v = 0; v < 8; ++v) {
                const int row = blockM + wm + i * 16 + v + (hi ? 8 : 0);
                float val = acc[i][j][v] + bv;
                if (ACT) val = fmaxf(val, 0.0f);
                if (cok) C[(size_t)row * N + col] = val;
            }
        }
    }
}

// ===========================================================================
// z = mu + eps * exp(0.5 * log_var)
// ===========================================================================
__global__ __launch_bounds__(256) void reparam_kernel(
    const float* __restrict__ mu, const float* __restrict__ lv,
    const float* __restrict__ eps, float* __restrict__ z, int n)
{
    int i = blockIdx.x * 256 + threadIdx.x;
    if (i < n) z[i] = mu[i] + eps[i] * __expf(0.5f * lv[i]);
}

__global__ void init_counter(unsigned int* c)
{
    if (threadIdx.x == 0 && blockIdx.x == 0) *c = 0u;
}

// ===========================================================================
// Reservoir scan: state_t = tanh(Wrec @ state_{t-1} + u_t), 8192 steps.
// 16 persistent workgroups; each owns 64 rows of Wrec resident in LDS
// (256KB <= 320KB CDNA5 LDS/WGP), preloaded by the Tensor Data Mover
// (tensor_load_to_lds + s_wait_tensorcnt). Per step: 4 lanes/row do 256-MAC
// LDS dots, wave32 shfl_xor reduce, tanh, publish, then device-scope counter
// sync (+ s_cluster_barrier fast path, NOP when not dispatched as a cluster).
// ===========================================================================
#define RC_BLOCKS 16
#define RC_ROWS   64   // N_DIM / RC_BLOCKS

__global__ __launch_bounds__(256) void reservoir_kernel(
    const float* __restrict__ Wrec, const float* __restrict__ u,
    float* __restrict__ hist, unsigned int* __restrict__ counter)
{
    extern __shared__ float lds[];
    float* Wl = lds;                          // RC_ROWS * N_DIM floats (256KB)
    float* st = lds + RC_ROWS * N_DIM;        // N_DIM floats (current state)

    const int tid  = threadIdx.x;
    const int b    = blockIdx.x;
    const int row  = tid >> 2;                // 0..63
    const int part = tid & 3;                 // 0..3
    const int grow = b * RC_ROWS + row;       // global state row

    // ---- Wrec slice -> LDS via Tensor Data Mover (one wave issues the DMA) ----
#if __has_builtin(__builtin_amdgcn_tensor_load_to_lds)
    if (tid < 32) {
        const unsigned ldsbase = (unsigned)__builtin_amdgcn_groupstaticsize();
        tdm_load_tile_2d_f32(ldsbase,
                             Wrec + (size_t)b * RC_ROWS * N_DIM,
                             N_DIM, RC_ROWS, N_DIM);
        __builtin_amdgcn_s_wait_tensorcnt(0);
    }
#else
    {
        const float* src = Wrec + (size_t)b * RC_ROWS * N_DIM;
        for (int i = tid * 4; i < RC_ROWS * N_DIM; i += 256 * 4)
            *reinterpret_cast<v4f*>(Wl + i) =
                *reinterpret_cast<const v4f*>(src + i);
    }
#endif
    // initial state = 0
    {
        int i = tid * 4;
        if (i < N_DIM) *reinterpret_cast<v4f*>(st + i) = v4f{0.f, 0.f, 0.f, 0.f};
    }
    __syncthreads();

    const float* wr = Wl + row * N_DIM + part * 256;
    const float* sp = st + part * 256;

    for (int t = 0; t < T_DIM; ++t) {
        float sum = 0.0f;
        #pragma unroll 8
        for (int k = 0; k < 256; ++k) sum = fmaf(wr[k], sp[k], sum);
        // reduce 4 partials (lanes row*4+part are consecutive within wave32)
        sum += __shfl_xor(sum, 1, 32);
        sum += __shfl_xor(sum, 2, 32);
        if (part == 0) {
            float val = tanhf(sum + u[(size_t)t * N_DIM + grow]);
            hist[(size_t)t * N_DIM + grow] = val;
        }
        __threadfence();
        __syncthreads();
        __builtin_amdgcn_s_cluster_barrier();   // cluster fast path (NOP uncl.)
        if (tid == 0) {
            __hip_atomic_fetch_add(counter, 1u, __ATOMIC_ACQ_REL,
                                   __HIP_MEMORY_SCOPE_AGENT);
            const unsigned target = (unsigned)RC_BLOCKS * (unsigned)(t + 1);
            while (__hip_atomic_load(counter, __ATOMIC_ACQUIRE,
                                     __HIP_MEMORY_SCOPE_AGENT) < target) {
                __builtin_amdgcn_s_sleep(1);
            }
        }
        __syncthreads();
        if (t + 1 < T_DIM) {   // pull the full new state into LDS
            int i = tid * 4;
            if (i < N_DIM)
                *reinterpret_cast<v4f*>(st + i) =
                    *reinterpret_cast<const v4f*>(hist + (size_t)t * N_DIM + i);
            __syncthreads();
        }
    }
}

// ===========================================================================
// Launch
// ===========================================================================
extern "C" void kernel_launch(void* const* d_in, const int* in_sizes, int n_in,
                              void* d_out, int out_size, void* d_ws, size_t ws_size,
                              hipStream_t stream)
{
    const float* data   = (const float*)d_in[0];
    // d_in[1] = label (unused by the outputs)
    const float* eps    = (const float*)d_in[2];
    const float* W_enc1 = (const float*)d_in[3];
    const float* b_enc1 = (const float*)d_in[4];
    const float* W_enc2 = (const float*)d_in[5];
    const float* b_enc2 = (const float*)d_in[6];
    const float* W_mu   = (const float*)d_in[7];
    const float* b_mu   = (const float*)d_in[8];
    const float* W_lv   = (const float*)d_in[9];
    const float* b_lv   = (const float*)d_in[10];
    const float* W_dec1 = (const float*)d_in[11];
    const float* b_dec1 = (const float*)d_in[12];
    const float* W_dec2 = (const float*)d_in[13];
    const float* b_dec2 = (const float*)d_in[14];
    const float* Win    = (const float*)d_in[15];
    const float* Wrec   = (const float*)d_in[16];
    const float* W_out  = (const float*)d_in[17];

    const size_t TH = (size_t)T_DIM * H_DIM;
    const size_t TD = (size_t)T_DIM * D_DIM;
    const size_t TN = (size_t)T_DIM * N_DIM;

    float* out     = (float*)d_out;
    float* encoded = out;                 // [T,H]
    float* mu      = out + TH;            // [T,H]
    float* lv      = out + 2 * TH;        // [T,H]
    float* z       = out + 3 * TH;        // [T,H]
    float* dec     = out + 4 * TH;        // [T,D]
    float* pred    = out + 4 * TH + TD;   // [T,C]

    float*        ws      = (float*)d_ws;
    float*        h1      = ws;                 // [T,512] scratch (reused)
    float*        u       = ws + TH;            // [T,N]
    float*        hist    = ws + TH + TN;       // [T,N]
    unsigned int* counter = (unsigned int*)(hist + TN);

    const dim3 blk(256);

    // VAE encoder
    gemm_wmma<1,1,0><<<dim3(T_DIM/128, H_DIM/64), blk, 0, stream>>>(
        data, W_enc1, b_enc1, h1, T_DIM, H_DIM, D_DIM);
    gemm_wmma<1,1,0><<<dim3(T_DIM/128, H_DIM/64), blk, 0, stream>>>(
        h1, W_enc2, b_enc2, encoded, T_DIM, H_DIM, H_DIM);
    gemm_wmma<0,1,0><<<dim3(T_DIM/128, H_DIM/64), blk, 0, stream>>>(
        encoded, W_mu, b_mu, mu, T_DIM, H_DIM, H_DIM);
    gemm_wmma<0,1,0><<<dim3(T_DIM/128, H_DIM/64), blk, 0, stream>>>(
        encoded, W_lv, b_lv, lv, T_DIM, H_DIM, H_DIM);
    reparam_kernel<<<dim3((unsigned)((TH + 255) / 256)), blk, 0, stream>>>(
        mu, lv, eps, z, (int)TH);

    // VAE decoder
    gemm_wmma<1,1,0><<<dim3(T_DIM/128, H_DIM/64), blk, 0, stream>>>(
        z, W_dec1, b_dec1, h1, T_DIM, H_DIM, H_DIM);
    gemm_wmma<0,1,0><<<dim3(T_DIM/128, D_DIM/64), blk, 0, stream>>>(
        h1, W_dec2, b_dec2, dec, T_DIM, D_DIM, H_DIM);

    // Reservoir inputs, scan, readout
    gemm_wmma<0,0,0><<<dim3(T_DIM/128, N_DIM/64), blk, 0, stream>>>(
        z, Win, nullptr, u, T_DIM, N_DIM, H_DIM);
    init_counter<<<1, 64, 0, stream>>>(counter);
    reservoir_kernel<<<dim3(RC_BLOCKS), blk,
                       (RC_ROWS * N_DIM + N_DIM) * sizeof(float), stream>>>(
        Wrec, u, hist, counter);
    gemm_wmma<0,0,1><<<dim3(T_DIM/128, 1), blk, 0, stream>>>(
        hist, W_out, nullptr, pred, T_DIM, C_DIM, N_DIM);
}
